// SpatialTR_22084721836077
// MI455X (gfx1250) — matrix operand order repaired and checked by reading, not compiled, not run on hardware
//
#include <hip/hip_runtime.h>
#include <hip/hip_bf16.h>
#include <math.h>

typedef __bf16 bf16_t;
typedef __attribute__((ext_vector_type(16))) __bf16 bfrag_t;
typedef __attribute__((ext_vector_type(8)))  __bf16 v8bf;
typedef __attribute__((ext_vector_type(8)))  float  v8f;

// ---------------- problem constants ----------------
#define NB   2
#define CCH  256
#define HH   96
#define WWF  96
#define HS   48           // strided h/w (and h_kv/w_kv)
#define POS  (HS*HS)      // 2304 positions
#define NG   8            // heads
#define DD   32           // head dim
#define MTOT (NB*POS)     // 4608 GEMM rows

// ---------------- small helpers ----------------
__global__ void __launch_bounds__(256) cvt_f32_bf16(const float* __restrict__ in,
                                                    bf16_t* __restrict__ out, int n) {
  int t = blockIdx.x * 256 + threadIdx.x;
  if (t < n) out[t] = (bf16_t)in[t];
}

// strided 2x downsample + layout transpose: x[n][c][96][96] -> xs[n][pos][c] bf16
__global__ void __launch_bounds__(256) ds_cvt_kernel(const float* __restrict__ x,
                                                     bf16_t* __restrict__ xs, int total) {
  int t = blockIdx.x * 256 + threadIdx.x;
  if (t >= total) return;
  int c = t & 255;
  int pos = (t >> 8) % POS;
  int nn = t / (POS * CCH);
  int hh = pos / HS, ww = pos % HS;
  xs[t] = (bf16_t)x[(((size_t)nn * CCH + c) * HH + hh * 2) * WWF + ww * 2];
}

// ---------------- generic NT bf16 WMMA GEMM ----------------
// D[m][ncol] = sum_k A[m][k] * B[ncol][k]   (A: MxK, B: NxK, both K-contiguous)
// mode 0: f32 row-major [M][N]
// mode 1: bf16 transposed [nb][N][Mper]   (nb = m / Mper, pos = m % Mper)
// mode 2: f32 transposed + bias[ncol]
__global__ void __launch_bounds__(256) gemm_nt_kernel(
    const bf16_t* __restrict__ A, const bf16_t* __restrict__ B,
    int K, int N, int tilesN, int mode, int Mper,
    float* __restrict__ outF, bf16_t* __restrict__ outT,
    const float* __restrict__ bias) {
  const int wid  = threadIdx.x >> 5;
  const int lane = threadIdx.x & 31;
  const int tile = blockIdx.x * 8 + wid;
  const int tm = tile / tilesN, tn = tile % tilesN;
  const int half = lane >> 4, col = lane & 15;

  const bf16_t* ap = A + (size_t)(tm * 16 + col) * K + half * 8;   // A row = lane%16
  const bf16_t* bp = B + (size_t)(tn * 16 + col) * K + half * 8;   // B col = lane%16
  v8f acc = {};
  for (int k = 0; k < K; k += 32) {
    v8bf a0 = *(const v8bf*)(ap + k);
    v8bf a1 = *(const v8bf*)(ap + k + 16);
    bfrag_t af;
#pragma unroll
    for (int i = 0; i < 8; i++) { af[i] = a0[i]; af[8 + i] = a1[i]; }
    v8bf b0 = *(const v8bf*)(bp + k);
    v8bf b1 = *(const v8bf*)(bp + k + 16);
    bfrag_t bf;
#pragma unroll
    for (int i = 0; i < 8; i++) { bf[i] = b0[i]; bf[8 + i] = b1[i]; }
    acc = __builtin_amdgcn_wmma_f32_16x16x32_bf16(false, af, false, bf,
                                                  (short)0, acc, false, false);
  }
  const int row0 = tm * 16 + half * 8;        // 8 consecutive output rows per lane
  const int ncol = tn * 16 + col;
  if (mode == 0) {
#pragma unroll
    for (int r = 0; r < 8; r++) outF[(size_t)(row0 + r) * N + ncol] = acc[r];
  } else if (mode == 1) {
    int nb = row0 / Mper, pos = row0 - nb * Mper;
    v8bf ov;
#pragma unroll
    for (int r = 0; r < 8; r++) ov[r] = (bf16_t)acc[r];
    *(v8bf*)(outT + ((size_t)(nb * N + ncol)) * Mper + pos) = ov;
  } else {
    float bv = bias[ncol];
    int nb = row0 / Mper, pos = row0 - nb * Mper;
    float* dst = outF + ((size_t)(nb * N + ncol)) * Mper + pos;
#pragma unroll
    for (int r = 0; r < 8; r++) dst[r] = acc[r] + bv;
  }
}

// ---------------- positional tables ----------------
// pos[g][i][j][dd] = inv_sqrt2 * sum_f emb(i,j,f) * Wm[g*32+dd][f]
// emb(f<64)=sin(diff/1000^(f/64)), emb(64+f)=cos(...), diff = 2*(i-j)
__global__ void __launch_bounds__(256) posembed_kernel(const float* __restrict__ Wm,
                                                       float* __restrict__ outp) {
  __shared__ float emb[128];
  const int bid = blockIdx.x;          // 0..2303
  const int i = bid / HS, j = bid % HS;
  const int t = threadIdx.x;
  if (t < 64) {
    float diff = 2.0f * (float)(i - j);
    // 1000^(-t/64) = exp2(-t * log2(1000)/64)
    float ang = diff * __builtin_amdgcn_exp2f(-(float)t * 0.15571537944784511f);
    emb[t]      = __sinf(ang);
    emb[64 + t] = __cosf(ang);
  }
  __syncthreads();
  const float* wrow = Wm + (size_t)t * 128;
  float s = 0.f;
#pragma unroll 8
  for (int f = 0; f < 128; f++) s += emb[f] * wrow[f];
  outp[(((size_t)(t >> 5) * HS + i) * HS + j) * DD + (t & 31)] = s * 0.70710678118654752f;
}

// e_key[n][g][pos] = sum_d bias[g*32+d] * Kt[n][g*32+d][pos]
__global__ void __launch_bounds__(256) ekey_kernel(const bf16_t* __restrict__ Kt,
                                                   const float* __restrict__ bias,
                                                   float* __restrict__ ek, int total) {
  int t = blockIdx.x * 256 + threadIdx.x;
  if (t >= total) return;
  int pos = t % POS;
  int gg = (t / POS) & 7;
  int nn = t / (POS * NG);
  const bf16_t* kp = Kt + ((size_t)(nn * CCH + gg * DD)) * POS + pos;
  const float* bb = bias + gg * DD;
  float s = 0.f;
#pragma unroll 8
  for (int d = 0; d < DD; d++) s += bb[d] * (float)kp[(size_t)d * POS];
  ek[t] = s;
}

// E[n][g][q][j] = sum_d Q[n][q][g*32+d] * pos[g][iq][j][d],  iq = isY ? q/48 : q%48
__global__ void __launch_bounds__(256) exy_kernel(const float* __restrict__ Q,
                                                  const float* __restrict__ pos,
                                                  float* __restrict__ E, int isY, int total) {
  int t = blockIdx.x * 256 + threadIdx.x;
  if (t >= total) return;
  int j = t % HS;
  int q = (t / HS) % POS;
  int gg = (t / (HS * POS)) & 7;
  int nn = t / (HS * POS * NG);
  int iq = isY ? (q / HS) : (q % HS);
  const float* qv = Q + ((size_t)(nn * POS + q)) * CCH + gg * DD;
  const float* pv = pos + (((size_t)(gg * HS + iq)) * HS + j) * DD;
  float s = 0.f;
#pragma unroll 8
  for (int d = 0; d < DD; d++) s += qv[d] * pv[d];
  E[t] = s;
}

// ---------------- flash attention core ----------------
// One wave per (n, g, 16-row q-tile). S tile built from E_key + E_x + E_y adds,
// online softmax in fp32, P->bf16 via wave-private LDS relayout, P.V via WMMA.
__global__ void __launch_bounds__(256) flash_kernel(
    const float* __restrict__ Ex, const float* __restrict__ Ey,
    const float* __restrict__ Ekey, const bf16_t* __restrict__ Vt,
    bf16_t* __restrict__ Ao) {
  __shared__ bf16_t pbuf[8 * 512];   // 16x32 bf16 per wave
  const int wid  = threadIdx.x >> 5;
  const int lane = threadIdx.x & 31;
  const int tile = blockIdx.x * 8 + wid;      // 0..2303
  const int qt = tile % (POS / 16);
  const int gg = (tile / (POS / 16)) & 7;
  const int nn = tile / ((POS / 16) * NG);
  const int qbase = qt * 16;
  const int half = lane >> 4, col = lane & 15;
  const int ng = nn * NG + gg;

  const float* exb = Ex + ((size_t)ng * POS + qbase + half * 8) * HS;
  const float* eyb = Ey + ((size_t)ng * POS + qbase + half * 8) * HS;
  const float* ekb = Ekey + (size_t)ng * POS;
  const bf16_t* vp0 = Vt + ((size_t)(nn * CCH + gg * DD + col)) * POS;       // d cols 0..15
  const bf16_t* vp1 = Vt + ((size_t)(nn * CCH + gg * DD + 16 + col)) * POS;  // d cols 16..31
  bf16_t* lds = pbuf + wid * 512;

  v8f o0 = {}, o1 = {};
  float rowM[8], rowL[8];
#pragma unroll
  for (int r = 0; r < 8; r++) { rowM[r] = -1e30f; rowL[r] = 0.f; }

  for (int kt = 0; kt < POS; kt += 32) {
    const int k0 = kt + col, k1 = k0 + 16;
    const int u0 = k0 / HS, v0 = k0 - u0 * HS;
    const int u1 = k1 / HS, v1 = k1 - u1 * HS;
    const float ek0 = ekb[k0], ek1 = ekb[k1];
    float p0[8], p1[8];
#pragma unroll
    for (int r = 0; r < 8; r++) {
      float s0 = ek0 + exb[r * HS + v0] + eyb[r * HS + u0];
      float s1 = ek1 + exb[r * HS + v1] + eyb[r * HS + u1];
      float mx = fmaxf(s0, s1);
#pragma unroll
      for (int off = 1; off < 16; off <<= 1) mx = fmaxf(mx, __shfl_xor(mx, off, 32));
      float mnew = fmaxf(rowM[r], mx);
      float alpha = __expf(rowM[r] - mnew);
      rowM[r] = mnew;
      float e0 = __expf(s0 - mnew);
      float e1 = __expf(s1 - mnew);
      float rs = e0 + e1;
#pragma unroll
      for (int off = 1; off < 16; off <<= 1) rs += __shfl_xor(rs, off, 32);
      rowL[r] = rowL[r] * alpha + rs;
      o0[r] *= alpha; o1[r] *= alpha;
      p0[r] = e0; p1[r] = e1;
    }
    // C-layout -> [m][k] LDS (wave-private; DS ops in-order within a wave)
#pragma unroll
    for (int r = 0; r < 8; r++) {
      int m = r + half * 8;
      lds[m * 32 + col]      = (bf16_t)p0[r];
      lds[m * 32 + 16 + col] = (bf16_t)p1[r];
    }
    // A-layout read back: row = lane%16, K contiguous 16 per lane half
    {
      int m = col;
      int klo = half * 16;
      v8bf alo = *(const v8bf*)(lds + m * 32 + klo);
      v8bf ahi = *(const v8bf*)(lds + m * 32 + klo + 8);
      bfrag_t af;
#pragma unroll
      for (int i = 0; i < 8; i++) { af[i] = alo[i]; af[8 + i] = ahi[i]; }
      int kk = kt + half * 16;
      bfrag_t b0 = *(const bfrag_t*)(vp0 + kk);
      bfrag_t b1 = *(const bfrag_t*)(vp1 + kk);
      o0 = __builtin_amdgcn_wmma_f32_16x16x32_bf16(false, af, false, b0,
                                                   (short)0, o0, false, false);
      o1 = __builtin_amdgcn_wmma_f32_16x16x32_bf16(false, af, false, b1,
                                                   (short)0, o1, false, false);
    }
  }
  // normalize + store attn output as [n][q][C] bf16 (feeds final NT GEMM)
#pragma unroll
  for (int r = 0; r < 8; r++) {
    int q = qbase + r + half * 8;
    float inv = 1.0f / rowL[r];
    bf16_t* dst = Ao + ((size_t)(nn * POS + q)) * CCH + gg * DD;
    dst[col]      = (bf16_t)(o0[r] * inv);
    dst[16 + col] = (bf16_t)(o1[r] * inv);
  }
}

// ---------------- bilinear 2x upsample + residual ----------------
__global__ void __launch_bounds__(256) resize_kernel(const float* __restrict__ out2,
                                                     const float* __restrict__ x,
                                                     const float* __restrict__ gamma,
                                                     float* __restrict__ out, int total) {
  int t = blockIdx.x * 256 + threadIdx.x;
  if (t >= total) return;
  int Wp = t % WWF;
  int Hp = (t / WWF) % HH;
  int c  = (t / (WWF * HH)) & 255;
  int nn = t / (WWF * HH * CCH);
  float fy = Hp * 0.5f - 0.25f, fx = Wp * 0.5f - 0.25f;
  int y0 = (int)floorf(fy), x0 = (int)floorf(fx);
  float dy = fy - (float)y0, dx = fx - (float)x0;
  int y0c = min(HS - 1, max(0, y0)), y1c = min(HS - 1, max(0, y0 + 1));
  int x0c = min(HS - 1, max(0, x0)), x1c = min(HS - 1, max(0, x0 + 1));
  const float* src = out2 + ((size_t)(nn * CCH + c)) * POS;
  float v00 = src[y0c * HS + x0c], v01 = src[y0c * HS + x1c];
  float v10 = src[y1c * HS + x0c], v11 = src[y1c * HS + x1c];
  float v = (1.f - dy) * ((1.f - dx) * v00 + dx * v01) +
            dy * ((1.f - dx) * v10 + dx * v11);
  out[t] = gamma[0] * v + x[t];
}

// ---------------- orchestration ----------------
extern "C" void kernel_launch(void* const* d_in, const int* in_sizes, int n_in,
                              void* d_out, int out_size, void* d_ws, size_t ws_size,
                              hipStream_t stream) {
  const float* x     = (const float*)d_in[0];
  const float* Wq    = (const float*)d_in[1];
  const float* Wk    = (const float*)d_in[2];
  const float* Wv    = (const float*)d_in[3];
  const float* Wx    = (const float*)d_in[4];
  const float* Wy    = (const float*)d_in[5];
  const float* abias = (const float*)d_in[6];
  const float* Wp    = (const float*)d_in[7];
  const float* bp    = (const float*)d_in[8];
  const float* gamma = (const float*)d_in[9];
  float* out = (float*)d_out;

  char* ws = (char*)d_ws;
  size_t off = 0;
  auto alloc = [&](size_t bytes) -> void* {
    void* p = ws + off;
    off += (bytes + 255) & ~(size_t)255;
    return p;
  };
  bf16_t* xs   = (bf16_t*)alloc((size_t)MTOT * CCH * 2);       // [n][pos][C]
  bf16_t* wqb  = (bf16_t*)alloc(CCH * CCH * 2);
  bf16_t* wkb  = (bf16_t*)alloc(CCH * CCH * 2);
  bf16_t* wvb  = (bf16_t*)alloc(CCH * CCH * 2);
  bf16_t* wpb  = (bf16_t*)alloc(CCH * CCH * 2);
  float*  Qb   = (float*) alloc((size_t)MTOT * CCH * 4);       // [n][pos][C]
  bf16_t* Kt   = (bf16_t*)alloc((size_t)MTOT * CCH * 2);       // [n][C][pos]
  bf16_t* Vt   = (bf16_t*)alloc((size_t)MTOT * CCH * 2);       // [n][C][pos]
  float*  posx = (float*) alloc((size_t)NG * HS * HS * DD * 4);
  float*  posy = (float*) alloc((size_t)NG * HS * HS * DD * 4);
  float*  ek   = (float*) alloc((size_t)NB * NG * POS * 4);
  float*  Ex   = (float*) alloc((size_t)NB * NG * POS * HS * 4);
  float*  Ey   = (float*) alloc((size_t)NB * NG * POS * HS * 4);
  bf16_t* Ao   = (bf16_t*)alloc((size_t)MTOT * CCH * 2);       // [n][pos][C]
  float*  out2 = (float*) alloc((size_t)MTOT * CCH * 4);       // [n][C][pos]

  // 1) weight conversions + input downsample/transpose
  cvt_f32_bf16<<<CCH * CCH / 256, 256, 0, stream>>>(Wq, wqb, CCH * CCH);
  cvt_f32_bf16<<<CCH * CCH / 256, 256, 0, stream>>>(Wk, wkb, CCH * CCH);
  cvt_f32_bf16<<<CCH * CCH / 256, 256, 0, stream>>>(Wv, wvb, CCH * CCH);
  cvt_f32_bf16<<<CCH * CCH / 256, 256, 0, stream>>>(Wp, wpb, CCH * CCH);
  {
    int total = MTOT * CCH;
    ds_cvt_kernel<<<(total + 255) / 256, 256, 0, stream>>>(x, xs, total);
  }

  // 2) Q/K/V projections (WMMA): 288x16 tiles, 8 waves/block
  const int tilesN = CCH / 16;
  const int gblk = (MTOT / 16) * tilesN / 8;  // 576
  gemm_nt_kernel<<<gblk, 256, 0, stream>>>(xs, wqb, CCH, CCH, tilesN, 0, POS, Qb, nullptr, nullptr);
  gemm_nt_kernel<<<gblk, 256, 0, stream>>>(xs, wkb, CCH, CCH, tilesN, 1, POS, nullptr, Kt, nullptr);
  gemm_nt_kernel<<<gblk, 256, 0, stream>>>(xs, wvb, CCH, CCH, tilesN, 1, POS, nullptr, Vt, nullptr);

  // 3) positional tables + energy components
  posembed_kernel<<<HS * HS, 256, 0, stream>>>(Wx, posx);
  posembed_kernel<<<HS * HS, 256, 0, stream>>>(Wy, posy);
  {
    int total = NB * NG * POS;
    ekey_kernel<<<(total + 255) / 256, 256, 0, stream>>>(Kt, abias, ek, total);
  }
  {
    int total = NB * NG * POS * HS;
    exy_kernel<<<(total + 255) / 256, 256, 0, stream>>>(Qb, posx, Ex, 0, total);
    exy_kernel<<<(total + 255) / 256, 256, 0, stream>>>(Qb, posy, Ey, 1, total);
  }

  // 4) flash attention (WMMA P.V): 2304 waves = 288 blocks
  flash_kernel<<<NB * NG * (POS / 16) / 8, 256, 0, stream>>>(Ex, Ey, ek, Vt, Ao);

  // 5) output projection + bias (WMMA), stored transposed [n][C][pos]
  gemm_nt_kernel<<<gblk, 256, 0, stream>>>(Ao, wpb, CCH, CCH, tilesN, 2, POS, out2, nullptr, bp);

  // 6) bilinear upsample + gamma residual
  {
    int total = NB * CCH * HH * WWF;
    resize_kernel<<<(total + 255) / 256, 256, 0, stream>>>(out2, x, gamma, out, total);
  }
}
